// MHSA_27470610826072
// MI455X (gfx1250) — matrix operand order repaired and checked by reading, not compiled, not run on hardware
//
#include <hip/hip_runtime.h>

typedef __bf16 bf16;
typedef __attribute__((ext_vector_type(8)))  bf16  v8bf;
typedef __attribute__((ext_vector_type(16))) bf16  v16bf;
typedef __attribute__((ext_vector_type(8)))  float v8f;
typedef __attribute__((ext_vector_type(4)))  int   v4i;

// ---------- CDNA5 async global->LDS (ASYNCcnt) with fallback ----------
#if defined(__has_builtin)
#if __has_builtin(__builtin_amdgcn_global_load_async_to_lds_b128)
#define HAVE_ASYNC_LDS 1
#endif
#endif
#ifndef HAVE_ASYNC_LDS
#define HAVE_ASYNC_LDS 0
#endif

#if HAVE_ASYNC_LDS
#if __has_builtin(__builtin_amdgcn_s_wait_asynccnt)
#define WAIT_ASYNC0() __builtin_amdgcn_s_wait_asynccnt(0)
#else
#define WAIT_ASYNC0() asm volatile("s_wait_asynccnt 0" ::: "memory")
#endif
static __device__ inline void async_copy16(const bf16* g, bf16* l) {
  typedef __attribute__((address_space(1))) v4i* gp4;  // global (prints as __device__)
  typedef __attribute__((address_space(3))) v4i* lp4;  // LDS
  // flat->global: same numeric address; flat->LDS: low 32 bits are the LDS offset
  __builtin_amdgcn_global_load_async_to_lds_b128(
      (gp4)(unsigned long long)g,
      (lp4)(unsigned int)(unsigned long long)l, 0, 0);
}
#endif

static __device__ inline v16bf frag_cat(v8bf lo, v8bf hi) {
  return __builtin_shufflevector(lo, hi, 0,1,2,3,4,5,6,7,8,9,10,11,12,13,14,15);
}

// A fragment 16x32 (MxK) from row-major A tile (leading dim ld, elements).
// lane l: m = l&15, h = l>>4; elems[0..7] -> K = h*8+i, elems[8..15] -> K = 16+h*8+i
static __device__ inline v16bf load_frag_a(const bf16* base, int ld) {
  const int lane = threadIdx.x & 31;
  const bf16* p = base + (lane & 15) * ld + (lane >> 4) * 8;
  v8bf lo = *(const v8bf*)(p);
  v8bf hi = *(const v8bf*)(p + 16);
  return frag_cat(lo, hi);
}

// B fragment 32x16 (KxN) from B^T stored row-major [n][k] (leading dim ld).
// lane l: n = l&15, h = l>>4; elems[0..7] -> K = h*8+i, elems[8..15] -> K = 16+h*8+i
static __device__ inline v16bf load_frag_b(const bf16* base, int ld) {
  const int lane = threadIdx.x & 31;
  const bf16* p = base + (lane & 15) * ld + (lane >> 4) * 8;
  v8bf lo = *(const v8bf*)(p);
  v8bf hi = *(const v8bf*)(p + 16);
  return frag_cat(lo, hi);
}

static __device__ inline v8f wmma_bf16(v16bf a, v16bf b, v8f c) {
  return __builtin_amdgcn_wmma_f32_16x16x32_bf16(false, a, false, b, (short)0, c,
                                                 false, false);
}

// ---------- VALU (DPP ROW_XMASK) butterfly reductions over 16-lane groups ----------
template <int XM>
static __device__ inline float dpp_xor16(float x) {
  return __int_as_float(__builtin_amdgcn_update_dpp(
      __float_as_int(x), __float_as_int(x), 0x160 | XM, 0xF, 0xF, false));
}
static __device__ inline float rowmax16(float x) {
  x = fmaxf(x, dpp_xor16<1>(x));
  x = fmaxf(x, dpp_xor16<2>(x));
  x = fmaxf(x, dpp_xor16<4>(x));
  x = fmaxf(x, dpp_xor16<8>(x));
  return x;
}
static __device__ inline float rowsum16(float x) {
  x += dpp_xor16<1>(x);
  x += dpp_xor16<2>(x);
  x += dpp_xor16<4>(x);
  x += dpp_xor16<8>(x);
  return x;
}

// ---------------- conversion kernels ----------------
__global__ void cvt_bf16(const float* __restrict__ in, bf16* __restrict__ out, int n) {
  int i = blockIdx.x * blockDim.x + threadIdx.x;
  if (i < n) out[i] = (bf16)in[i];
}

// in fp32 [rows][cols] -> out bf16 [cols][rows]  (pre-transpose weights)
__global__ void cvt_bf16_t(const float* __restrict__ in, bf16* __restrict__ out,
                           int rows, int cols) {
  int i = blockIdx.x * blockDim.x + threadIdx.x;
  if (i < rows * cols) {
    int c = i / rows, r = i - c * rows;
    out[i] = (bf16)in[(size_t)r * cols + c];
  }
}

// ---------------- bf16 WMMA GEMM: C[M,N] = A[M,K] * Bt[N,K]^T ----------------
// 256 thr (8 waves), block tile 128x128, wave tile 32x64, K-step 32,
// double-buffered LDS, one barrier per K-step, async global->LDS when available.
template <typename OutT>
__global__ __launch_bounds__(256) void gemm_bf16(const bf16* __restrict__ A,
                                                 const bf16* __restrict__ Bt,
                                                 OutT* __restrict__ C,
                                                 int M, int N, int K) {
  __shared__ __align__(16) bf16 As[2][128 * 32];
  __shared__ __align__(16) bf16 Bs[2][128 * 32];
  const int t = threadIdx.x;
  const int m0 = blockIdx.y * 128;
  const int n0 = blockIdx.x * 128;
  const int wave = t >> 5, lane = t & 31;
  const int wm = wave & 3, wn = wave >> 2;
  // two vec8 staging slots per thread (512 slots per 128x32 tile)
  const int r0 = t >> 2, c0 = (t & 3) * 8;
  const int r1 = (t + 256) >> 2, c1 = ((t + 256) & 3) * 8;

  v8f zero = {};
  v8f acc[2][4];
#pragma unroll
  for (int mi = 0; mi < 2; mi++)
#pragma unroll
    for (int ni = 0; ni < 4; ni++) acc[mi][ni] = zero;

  const int NK = K >> 5;

#if HAVE_ASYNC_LDS
  async_copy16(&A[(size_t)(m0 + r0) * K + c0], &As[0][r0 * 32 + c0]);
  async_copy16(&A[(size_t)(m0 + r1) * K + c1], &As[0][r1 * 32 + c1]);
  async_copy16(&Bt[(size_t)(n0 + r0) * K + c0], &Bs[0][r0 * 32 + c0]);
  async_copy16(&Bt[(size_t)(n0 + r1) * K + c1], &Bs[0][r1 * 32 + c1]);
  WAIT_ASYNC0();
  __syncthreads();
  for (int kt = 0; kt < NK; ++kt) {
    const int cur = kt & 1, nxt = cur ^ 1;
    if (kt + 1 < NK) {
      const int k0 = (kt + 1) << 5;
      async_copy16(&A[(size_t)(m0 + r0) * K + k0 + c0], &As[nxt][r0 * 32 + c0]);
      async_copy16(&A[(size_t)(m0 + r1) * K + k0 + c1], &As[nxt][r1 * 32 + c1]);
      async_copy16(&Bt[(size_t)(n0 + r0) * K + k0 + c0], &Bs[nxt][r0 * 32 + c0]);
      async_copy16(&Bt[(size_t)(n0 + r1) * K + k0 + c1], &Bs[nxt][r1 * 32 + c1]);
    }
    v16bf af[2], bfr[4];
#pragma unroll
    for (int mi = 0; mi < 2; mi++)
      af[mi] = load_frag_a(&As[cur][(wm * 32 + mi * 16) * 32], 32);
#pragma unroll
    for (int ni = 0; ni < 4; ni++)
      bfr[ni] = load_frag_b(&Bs[cur][(wn * 64 + ni * 16) * 32], 32);
#pragma unroll
    for (int mi = 0; mi < 2; mi++)
#pragma unroll
      for (int ni = 0; ni < 4; ni++)
        acc[mi][ni] = wmma_bf16(af[mi], bfr[ni], acc[mi][ni]);
    if (kt + 1 < NK) { WAIT_ASYNC0(); __syncthreads(); }
  }
#else
  {
    v8bf a0 = *(const v8bf*)&A[(size_t)(m0 + r0) * K + c0];
    v8bf a1 = *(const v8bf*)&A[(size_t)(m0 + r1) * K + c1];
    v8bf b0 = *(const v8bf*)&Bt[(size_t)(n0 + r0) * K + c0];
    v8bf b1 = *(const v8bf*)&Bt[(size_t)(n0 + r1) * K + c1];
    *(v8bf*)&As[0][r0 * 32 + c0] = a0;
    *(v8bf*)&As[0][r1 * 32 + c1] = a1;
    *(v8bf*)&Bs[0][r0 * 32 + c0] = b0;
    *(v8bf*)&Bs[0][r1 * 32 + c1] = b1;
  }
  __syncthreads();
  for (int kt = 0; kt < NK; ++kt) {
    const int cur = kt & 1, nxt = cur ^ 1;
    v8bf a0, a1, b0, b1;
    if (kt + 1 < NK) {
      const int k0 = (kt + 1) << 5;
      a0 = *(const v8bf*)&A[(size_t)(m0 + r0) * K + k0 + c0];
      a1 = *(const v8bf*)&A[(size_t)(m0 + r1) * K + k0 + c1];
      b0 = *(const v8bf*)&Bt[(size_t)(n0 + r0) * K + k0 + c0];
      b1 = *(const v8bf*)&Bt[(size_t)(n0 + r1) * K + k0 + c1];
    }
    v16bf af[2], bfr[4];
#pragma unroll
    for (int mi = 0; mi < 2; mi++)
      af[mi] = load_frag_a(&As[cur][(wm * 32 + mi * 16) * 32], 32);
#pragma unroll
    for (int ni = 0; ni < 4; ni++)
      bfr[ni] = load_frag_b(&Bs[cur][(wn * 64 + ni * 16) * 32], 32);
#pragma unroll
    for (int mi = 0; mi < 2; mi++)
#pragma unroll
      for (int ni = 0; ni < 4; ni++)
        acc[mi][ni] = wmma_bf16(af[mi], bfr[ni], acc[mi][ni]);
    if (kt + 1 < NK) {
      *(v8bf*)&As[nxt][r0 * 32 + c0] = a0;
      *(v8bf*)&As[nxt][r1 * 32 + c1] = a1;
      *(v8bf*)&Bs[nxt][r0 * 32 + c0] = b0;
      *(v8bf*)&Bs[nxt][r1 * 32 + c1] = b1;
      __syncthreads();
    }
  }
#endif

  const int half = lane >> 4, nl = lane & 15;
#pragma unroll
  for (int mi = 0; mi < 2; mi++)
#pragma unroll
    for (int ni = 0; ni < 4; ni++) {
      int mg = m0 + wm * 32 + mi * 16 + half * 8;
      int ng = n0 + wn * 64 + ni * 16 + nl;
#pragma unroll
      for (int r = 0; r < 8; r++)
        C[(size_t)(mg + r) * N + ng] = (OutT)acc[mi][ni][r];
    }
}

// ---------------- flash attention ----------------
// grid: x = query chunk (16 of 128 rows), y = bh (128). block = 256 thr (8 waves).
// qkv: bf16 [8192][3072]; out: bf16 [8192][1024] (token-major, col = h*64+d)
__global__ __launch_bounds__(256) void attn_fa(const bf16* __restrict__ qkv,
                                               bf16* __restrict__ o_out) {
  const int qc = blockIdx.x;
  const int bh = blockIdx.y;
  const int b = bh >> 4, h = bh & 15;
  const int t = threadIdx.x, lane = t & 31, wave = t >> 5;
  const int half = lane >> 4, nl = lane & 15;

  __shared__ __align__(16) bf16 Qs[128 * 64];     // 16 KB (pre-scaled by 1/32)
  __shared__ __align__(16) bf16 Ks[2][64 * 64];   // 16 KB
  __shared__ __align__(16) bf16 Vt[2][64 * 64];   // 16 KB (V^T: [d][key])
  __shared__ __align__(16) bf16 Ps[8 * 16 * 64];  // 16 KB per-wave P staging

  const size_t row_base = (size_t)b * 1024;
  // two vec8 staging slots per thread for 64x64 tiles (512 slots)
  const int sr0 = t >> 3, sc0 = (t & 7) * 8;
  const int sr1 = (t + 256) >> 3, sc1 = ((t + 256) & 7) * 8;

  // stage Q chunk [128 x 64], pre-scaled by 1/sqrt(1024)
#pragma unroll
  for (int i = 0; i < 4; i++) {
    int idx = t + i * 256;
    int row = idx >> 3;
    int c = (idx & 7) * 8;
    v8bf v = *(const v8bf*)&qkv[(row_base + qc * 128 + row) * 3072 + h * 64 + c];
    v8bf q;
#pragma unroll
    for (int j = 0; j < 8; j++) q[j] = (bf16)((float)v[j] * 0.03125f);
    *(v8bf*)&Qs[row * 64 + c] = q;
  }

  // stage K/V tile 0
#if HAVE_ASYNC_LDS
  async_copy16(&qkv[(row_base + sr0) * 3072 + 1024 + h * 64 + sc0], &Ks[0][sr0 * 64 + sc0]);
  async_copy16(&qkv[(row_base + sr1) * 3072 + 1024 + h * 64 + sc1], &Ks[0][sr1 * 64 + sc1]);
#else
  {
    v8bf k0 = *(const v8bf*)&qkv[(row_base + sr0) * 3072 + 1024 + h * 64 + sc0];
    v8bf k1 = *(const v8bf*)&qkv[(row_base + sr1) * 3072 + 1024 + h * 64 + sc1];
    *(v8bf*)&Ks[0][sr0 * 64 + sc0] = k0;
    *(v8bf*)&Ks[0][sr1 * 64 + sc1] = k1;
  }
#endif
  {
    v8bf v0 = *(const v8bf*)&qkv[(row_base + sr0) * 3072 + 2048 + h * 64 + sc0];
    v8bf v1 = *(const v8bf*)&qkv[(row_base + sr1) * 3072 + 2048 + h * 64 + sc1];
#pragma unroll
    for (int j = 0; j < 8; j++) Vt[0][(sc0 + j) * 64 + sr0] = v0[j];
#pragma unroll
    for (int j = 0; j < 8; j++) Vt[0][(sc1 + j) * 64 + sr1] = v1[j];
  }
#if HAVE_ASYNC_LDS
  WAIT_ASYNC0();
#endif
  __syncthreads();

  v16bf qf[2];
  qf[0] = load_frag_a(&Qs[(wave * 16) * 64 + 0], 64);
  qf[1] = load_frag_a(&Qs[(wave * 16) * 64 + 32], 64);

  v8f zero = {};
  float m_run[8], l_run[8];  // l_run: per-lane partial (sum deferred to epilogue)
  v8f o[4];
#pragma unroll
  for (int r = 0; r < 8; r++) { m_run[r] = -1e30f; l_run[r] = 0.f; }
#pragma unroll
  for (int di = 0; di < 4; di++) o[di] = zero;

  for (int kt = 0; kt < 16; ++kt) {
    const int cur = kt & 1, nxt = cur ^ 1;
    v8bf vr0, vr1;
#if !HAVE_ASYNC_LDS
    v8bf kr0, kr1;
#endif
    if (kt + 1 < 16) {
      const size_t krow = row_base + (kt + 1) * 64;
#if HAVE_ASYNC_LDS
      async_copy16(&qkv[(krow + sr0) * 3072 + 1024 + h * 64 + sc0], &Ks[nxt][sr0 * 64 + sc0]);
      async_copy16(&qkv[(krow + sr1) * 3072 + 1024 + h * 64 + sc1], &Ks[nxt][sr1 * 64 + sc1]);
#else
      kr0 = *(const v8bf*)&qkv[(krow + sr0) * 3072 + 1024 + h * 64 + sc0];
      kr1 = *(const v8bf*)&qkv[(krow + sr1) * 3072 + 1024 + h * 64 + sc1];
#endif
      vr0 = *(const v8bf*)&qkv[(krow + sr0) * 3072 + 2048 + h * 64 + sc0];
      vr1 = *(const v8bf*)&qkv[(krow + sr1) * 3072 + 2048 + h * 64 + sc1];
    }

    // S = Q * K^T  (wave tile: 16 q x 64 keys), Q pre-scaled
    v8f s[4];
#pragma unroll
    for (int ni = 0; ni < 4; ni++) s[ni] = zero;
#pragma unroll
    for (int ks = 0; ks < 2; ks++)
#pragma unroll
      for (int ni = 0; ni < 4; ni++) {
        v16bf bk = load_frag_b(&Ks[cur][(ni * 16) * 64 + ks * 32], 64);
        s[ni] = wmma_bf16(qf[ks], bk, s[ni]);
      }

    // online softmax: VALU-only row max (DPP xmask), per-lane partial row sum
#pragma unroll
    for (int r = 0; r < 8; r++) {
      float s0 = s[0][r], s1 = s[1][r], s2 = s[2][r], s3 = s[3][r];
      float tm = rowmax16(fmaxf(fmaxf(s0, s1), fmaxf(s2, s3)));
      float mn = fmaxf(m_run[r], tm);
      float alpha = __expf(m_run[r] - mn);
      m_run[r] = mn;
      float p0 = __expf(s0 - mn), p1 = __expf(s1 - mn);
      float p2 = __expf(s2 - mn), p3 = __expf(s3 - mn);
      s[0][r] = p0; s[1][r] = p1; s[2][r] = p2; s[3][r] = p3;
      l_run[r] = l_run[r] * alpha + (p0 + p1 + p2 + p3);
#pragma unroll
      for (int di = 0; di < 4; di++) o[di][r] *= alpha;
    }

    // C-layout -> A-layout: round-trip P through per-wave LDS region
    bf16* pb = &Ps[wave * 16 * 64];
#pragma unroll
    for (int ni = 0; ni < 4; ni++)
#pragma unroll
      for (int r = 0; r < 8; r++)
        pb[(half * 8 + r) * 64 + ni * 16 + nl] = (bf16)s[ni][r];
    v16bf pf0 = load_frag_a(pb + 0, 64);
    v16bf pf1 = load_frag_a(pb + 32, 64);

    // O += P * V
#pragma unroll
    for (int di = 0; di < 4; di++) {
      v16bf b0 = load_frag_b(&Vt[cur][(di * 16) * 64 + 0], 64);
      o[di] = wmma_bf16(pf0, b0, o[di]);
    }
#pragma unroll
    for (int di = 0; di < 4; di++) {
      v16bf b1 = load_frag_b(&Vt[cur][(di * 16) * 64 + 32], 64);
      o[di] = wmma_bf16(pf1, b1, o[di]);
    }

    if (kt + 1 < 16) {
#pragma unroll
      for (int j = 0; j < 8; j++) Vt[nxt][(sc0 + j) * 64 + sr0] = vr0[j];
#pragma unroll
      for (int j = 0; j < 8; j++) Vt[nxt][(sc1 + j) * 64 + sr1] = vr1[j];
#if !HAVE_ASYNC_LDS
      *(v8bf*)&Ks[nxt][sr0 * 64 + sc0] = kr0;
      *(v8bf*)&Ks[nxt][sr1 * 64 + sc1] = kr1;
#else
      WAIT_ASYNC0();
#endif
      __syncthreads();
    }
  }

  // deferred denominator reduction (one butterfly per row, after all tiles)
#pragma unroll
  for (int r = 0; r < 8; r++) l_run[r] = rowsum16(l_run[r]);

  // normalize + store (token-major [8192][1024], col = h*64 + d)
#pragma unroll
  for (int di = 0; di < 4; di++) {
    int col = h * 64 + di * 16 + nl;
#pragma unroll
    for (int r = 0; r < 8; r++) {
      size_t rowg = row_base + qc * 128 + wave * 16 + half * 8 + r;
      o_out[rowg * 1024 + col] = (bf16)(o[di][r] / l_run[r]);
    }
  }
}

// ---------------- launcher ----------------
extern "C" void kernel_launch(void* const* d_in, const int* in_sizes, int n_in,
                              void* d_out, int out_size, void* d_ws, size_t ws_size,
                              hipStream_t stream) {
  const float* x     = (const float*)d_in[0];  // [8,1024,1024]
  const float* w_qkv = (const float*)d_in[1];  // [1024,3072]
  const float* w_out = (const float*)d_in[2];  // [1024,1024]
  float* out = (float*)d_out;                  // [8,1024,1024]

  char* ws = (char*)d_ws;
  size_t off = 0;
  bf16* x_bf   = (bf16*)(ws + off); off += (size_t)8192 * 1024 * 2;  // 16 MB
  bf16* wqkv_t = (bf16*)(ws + off); off += (size_t)3072 * 1024 * 2;  //  6 MB
  bf16* wout_t = (bf16*)(ws + off); off += (size_t)1024 * 1024 * 2;  //  2 MB
  bf16* qkv    = (bf16*)(ws + off); off += (size_t)8192 * 3072 * 2;  // 48 MB
  bf16* aout   = (bf16*)(ws + off); off += (size_t)8192 * 1024 * 2;  // 16 MB

  cvt_bf16<<<(8192 * 1024) / 256, 256, 0, stream>>>(x, x_bf, 8192 * 1024);
  cvt_bf16_t<<<(3072 * 1024) / 256, 256, 0, stream>>>(w_qkv, wqkv_t, 1024, 3072);
  cvt_bf16_t<<<(1024 * 1024) / 256, 256, 0, stream>>>(w_out, wout_t, 1024, 1024);

  // QKV projection: [8192,1024] x [1024,3072] -> bf16 [8192,3072]
  gemm_bf16<bf16><<<dim3(3072 / 128, 8192 / 128), 256, 0, stream>>>(
      x_bf, wqkv_t, qkv, 8192, 3072, 1024);

  // attention -> bf16 [8192,1024]
  attn_fa<<<dim3(16, 128), 256, 0, stream>>>(qkv, aout);

  // output projection: [8192,1024] x [1024,1024] -> fp32 d_out
  gemm_bf16<float><<<dim3(1024 / 128, 8192 / 128), 256, 0, stream>>>(
      aout, wout_t, out, 8192, 1024, 1024);
}
